// att_res_mlp_LN2_27127013441731
// MI455X (gfx1250) — compile-verified, matrix operand order checked by reading, not run on hardware
//
#include <hip/hip_runtime.h>
#include <hip/hip_bf16.h>

typedef _Float16 v16h __attribute__((ext_vector_type(16)));
typedef _Float16 v8h  __attribute__((ext_vector_type(8)));
typedef _Float16 h4   __attribute__((ext_vector_type(4)));
typedef float    v8f  __attribute__((ext_vector_type(8)));
typedef unsigned int u32x4 __attribute__((ext_vector_type(4)));
typedef int          i32x4 __attribute__((ext_vector_type(4)));
typedef int          i32x8 __attribute__((ext_vector_type(8)));

#define DEV __device__ __forceinline__

#if defined(__has_builtin)
#if __has_builtin(__builtin_amdgcn_tensor_load_to_lds)
#define HAVE_TDM 1
#endif
#if __has_builtin(__builtin_amdgcn_s_wait_tensorcnt)
#define HAVE_TCNT 1
#endif
#endif

// ---------- per-lane helpers ----------
template<int D>
DEV void ln_(float* x, const float* g, const float* b) {
    float m = 0.f;
#pragma unroll
    for (int i = 0; i < D; ++i) m += x[i];
    m *= (1.0f / D);
    float v = 0.f;
#pragma unroll
    for (int i = 0; i < D; ++i) { float d = x[i] - m; v += d * d; }
    v *= (1.0f / D);
    float r = rsqrtf(v + 1e-5f);
#pragma unroll
    for (int i = 0; i < D; ++i) x[i] = (x[i] - m) * r * g[i] + b[i];
}

template<int D>
DEV void relu_(float* x) {
#pragma unroll
    for (int i = 0; i < D; ++i) x[i] = fmaxf(x[i], 0.f);
}

template<int O, int I>
DEV void matvec(const float* W, const float* x, const float* bias, float* y) {
#pragma unroll
    for (int o = 0; o < O; ++o) {
        float s = bias ? bias[o] : 0.f;
#pragma unroll
        for (int i = 0; i < I; ++i) s += W[o * I + i] * x[i];
        y[o] = s;
    }
}

// ---------- Tensor Data Mover: async rows x cols f32 tile, global -> LDS ----------
DEV void tdm_load_tile(const float* gsrc, void* ldst, int cols, int rows) {
#ifdef HAVE_TDM
    unsigned long long ga = (unsigned long long)(uintptr_t)gsrc;
    unsigned ldsOff = (unsigned)(uintptr_t)ldst;          // low 32 bits = LDS offset
    u32x4 g0 = { 1u,                                      // count=1, user D#
                 ldsOff,                                  // lds_addr
                 (unsigned)(ga & 0xFFFFFFFFu),            // global_addr[31:0]
                 (unsigned)((ga >> 32) & 0x01FFFFFFu) | (2u << 30) }; // ga[56:32]|type=2
    i32x8 g1 = { (int)(2u << 16),                         // data_size = 4B
                 (int)((unsigned)cols << 16),             // tensor_dim0
                 (int)((unsigned)rows << 16),             // tensor_dim1
                 (int)((unsigned)cols << 16),             // tile_dim0
                 rows,                                    // tile_dim1 (tile_dim2=0)
                 cols,                                    // tensor_dim0_stride
                 0, 0 };
    i32x4 gz = { 0, 0, 0, 0 };
#if __has_include(<hip/amd_detail/amd_gfx1250_TDM.h>)
    i32x8 gz8 = { 0, 0, 0, 0, 0, 0, 0, 0 };               // 6-arg toolchain
    __builtin_amdgcn_tensor_load_to_lds(g0, g1, gz, gz, gz8, 0);
#else
    __builtin_amdgcn_tensor_load_to_lds(g0, g1, gz, gz, 0); // 5-arg ROCm 7.2
#endif
#else
    const int lane = threadIdx.x & 31;
    float* d = (float*)ldst;
    const int n4 = (cols * rows) >> 2;
    for (int i = lane; i < n4; i += 32)
        *reinterpret_cast<float4*>(d + i * 4) =
            *reinterpret_cast<const float4*>(gsrc + i * 4);
#endif
}

DEV void tdm_wait(int n) {
#if defined(HAVE_TDM) && defined(HAVE_TCNT)
    if (n == 0)      __builtin_amdgcn_s_wait_tensorcnt(0);
    else if (n == 2) __builtin_amdgcn_s_wait_tensorcnt(2);
    else             __builtin_amdgcn_s_wait_tensorcnt(1);
#endif
}

// ---------- parameters ----------
struct KParams {
    const float *kv, *q;
    const float *Wk, *bk, *Wv, *bv;
    const float *Wq1, *gq, *bq, *Wq2, *bq2;
    const float *Wr1, *br1, *gr1, *br1n, *Wr2, *br2;
    const float *gr2, *br2n, *Wr3, *br3;
    const float *g11, *b11, *W11, *g12, *b12, *W12, *g13, *b13;
    const float *W21, *g21, *b21, *W22, *g22, *b22;
    const float *W31, *g31, *b31, *W32, *g32, *b32;
    const float *W41, *g41, *b41, *W42, *b42;
    const float *gf1, *bf1, *Wf1, *gf2, *bf2, *Wf2, *bf2b;
    float* out;
};

// LDS offsets (floats) for the trunk parameter pool
enum : int {
    O_bk = 0, O_bv = O_bk + 32, O_br1 = O_bv + 32, O_Wr1q = O_br1 + 32,
    O_Wq1 = O_Wr1q + 256, O_gq = O_Wq1 + 128, O_bq = O_gq + 16, O_Wq2 = O_bq + 16,
    O_bq2 = O_Wq2 + 512, O_gr1 = O_bq2 + 32, O_br1n = O_gr1 + 32, O_Wr2 = O_br1n + 32,
    O_br2 = O_Wr2 + 512, O_gr2 = O_br2 + 16, O_br2n = O_gr2 + 32, O_Wr3 = O_br2n + 32,
    O_br3 = O_Wr3 + 512,
    O_g11 = O_br3 + 16, O_b11 = O_g11 + 16, O_W11 = O_b11 + 16, O_g12 = O_W11 + 256,
    O_b12 = O_g12 + 16, O_W12 = O_b12 + 16, O_g13 = O_W12 + 256, O_b13 = O_g13 + 16,
    O_W21 = O_b13 + 16, O_g21 = O_W21 + 256, O_b21 = O_g21 + 16, O_W22 = O_b21 + 16,
    O_g22 = O_W22 + 256, O_b22 = O_g22 + 16,
    O_W31 = O_b22 + 16, O_g31 = O_W31 + 256, O_b31 = O_g31 + 16, O_W32 = O_b31 + 16,
    O_g32 = O_W32 + 256, O_b32 = O_g32 + 16,
    O_W41 = O_b32 + 16, O_g41 = O_W41 + 256, O_b41 = O_g41 + 16, O_W42 = O_b41 + 16,
    O_b42 = O_W42 + 256,
    O_gf1 = O_b42 + 16, O_bf1 = O_gf1 + 48, O_Wf1 = O_bf1 + 48, O_gf2 = O_Wf1 + 1152,
    O_bf2 = O_gf2 + 24, O_Wf2 = O_bf2 + 24, O_bf2b = O_Wf2 + 192, O_TOT = O_bf2b + 8
};

#define SHUF16(lo, hi) __builtin_shufflevector(lo, hi, 0,1,2,3,4,5,6,7,8,9,10,11,12,13,14,15)

// 128 threads = 4 waves, persistent blocks, per-wave TDM double-buffered pipeline.
// All 24 loop-invariant B fragments (192 VGPRs) are pinned in the wave32 1024-VGPR
// file, so the steady-state WMMA phase has no B-side LDS traffic or dscnt waits.
__global__ __launch_bounds__(128) void fused_net(KParams p, int numTiles) {
    __shared__ __attribute__((aligned(16))) float    sF32[2][128 * 128]; // TDM kv landing
    __shared__ __attribute__((aligned(16))) float    sQ  [2][128 * 8];   // TDM q landing
    __shared__ __attribute__((aligned(16))) _Float16 sWk [32 * 128];
    __shared__ __attribute__((aligned(16))) _Float16 sWv [32 * 128];
    __shared__ __attribute__((aligned(16))) _Float16 sWr1[32 * 128];
    __shared__ __attribute__((aligned(16))) _Float16 sA  [128 * 128];    // f16 A operands
    __shared__ __attribute__((aligned(16))) float    sK  [128 * 32];
    __shared__ __attribute__((aligned(16))) float    sV  [128 * 32];
    __shared__ __attribute__((aligned(16))) float    sR1 [128 * 32];
    __shared__ __attribute__((aligned(16))) float    sP  [O_TOT];        // trunk params

    const int tid  = threadIdx.x;
    const int lane = tid & 31;
    const int wave = __builtin_amdgcn_readfirstlane(tid >> 5);
    const int wr0  = wave * 32;                 // wave's first block-local row
    const int t0   = blockIdx.x;

    // prologue: kick off DMA of the first tile before anything else
    if (t0 < numTiles) {
        tdm_load_tile(p.kv + ((size_t)t0 * 128 + wr0) * 128, &sF32[0][wr0 * 128], 128, 32);
        tdm_load_tile(p.q  + ((size_t)t0 * 128 + wr0) * 8,   &sQ  [0][wr0 * 8],     8, 32);
    }

    // one-time staging: f16 weights for the K=128 GEMMs
    for (int i = tid; i < 32 * 128; i += 128) {
        sWk[i] = (_Float16)p.Wk[i];
        sWv[i] = (_Float16)p.Wv[i];
        int n = i >> 7, k = i & 127;            // Wr1 rows have stride 136
        sWr1[i] = (_Float16)p.Wr1[n * 136 + k];
    }
    // one-time staging: trunk parameter pool (uniform LDS reads broadcast to all lanes)
#define CPY(off, src, n) for (int i = tid; i < (n); i += 128) sP[(off) + i] = (src)[i];
    CPY(O_bk, p.bk, 32)   CPY(O_bv, p.bv, 32)   CPY(O_br1, p.br1, 32)
    for (int i = tid; i < 256; i += 128) sP[O_Wr1q + i] = p.Wr1[(i >> 3) * 136 + 128 + (i & 7)];
    CPY(O_Wq1, p.Wq1, 128) CPY(O_gq, p.gq, 16)  CPY(O_bq, p.bq, 16)
    CPY(O_Wq2, p.Wq2, 512) CPY(O_bq2, p.bq2, 32)
    CPY(O_gr1, p.gr1, 32)  CPY(O_br1n, p.br1n, 32)
    CPY(O_Wr2, p.Wr2, 512) CPY(O_br2, p.br2, 16)
    CPY(O_gr2, p.gr2, 32)  CPY(O_br2n, p.br2n, 32)
    CPY(O_Wr3, p.Wr3, 512) CPY(O_br3, p.br3, 16)
    CPY(O_g11, p.g11, 16)  CPY(O_b11, p.b11, 16) CPY(O_W11, p.W11, 256)
    CPY(O_g12, p.g12, 16)  CPY(O_b12, p.b12, 16) CPY(O_W12, p.W12, 256)
    CPY(O_g13, p.g13, 16)  CPY(O_b13, p.b13, 16)
    CPY(O_W21, p.W21, 256) CPY(O_g21, p.g21, 16) CPY(O_b21, p.b21, 16)
    CPY(O_W22, p.W22, 256) CPY(O_g22, p.g22, 16) CPY(O_b22, p.b22, 16)
    CPY(O_W31, p.W31, 256) CPY(O_g31, p.g31, 16) CPY(O_b31, p.b31, 16)
    CPY(O_W32, p.W32, 256) CPY(O_g32, p.g32, 16) CPY(O_b32, p.b32, 16)
    CPY(O_W41, p.W41, 256) CPY(O_g41, p.g41, 16) CPY(O_b41, p.b41, 16)
    CPY(O_W42, p.W42, 256) CPY(O_b42, p.b42, 16)
    CPY(O_gf1, p.gf1, 48)  CPY(O_bf1, p.bf1, 48) CPY(O_Wf1, p.Wf1, 1152)
    CPY(O_gf2, p.gf2, 24)  CPY(O_bf2, p.bf2, 24) CPY(O_Wf2, p.Wf2, 192)
    CPY(O_bf2b, p.bf2b, 8)
#undef CPY
    __syncthreads();

    const int lm = lane & 15;                   // M (A) / N (B) within tile
    const int kh = (lane >> 4) << 3;            // K half select
    const int m0 = (lane >> 4) << 3;            // C/D: vgpr j -> row j + m0

    // ---- pin all loop-invariant B fragments in registers (3 x 2 x 4 x v16h) ----
    v16h Bfr[3][2][4];
    {
        const _Float16* Ws[3] = { sWk, sWv, sWr1 };
#pragma unroll
        for (int mm = 0; mm < 3; ++mm)
#pragma unroll
            for (int nt = 0; nt < 2; ++nt)
#pragma unroll
                for (int kt = 0; kt < 4; ++kt) {
                    const _Float16* bp = &Ws[mm][(nt * 16 + lm) * 128 + kt * 32 + kh];
                    v8h lo = *(const v8h*)bp;
                    v8h hi = *(const v8h*)(bp + 16);
                    Bfr[mm][nt][kt] = SHUF16(lo, hi);
                }
    }

    int pipe = 0;
    for (int t = t0; t < numTiles; t += gridDim.x, pipe ^= 1) {
        const int tn = t + (int)gridDim.x;
        if (tn < numTiles) {                    // prefetch next tile, overlap with compute
            tdm_load_tile(p.kv + ((size_t)tn * 128 + wr0) * 128, &sF32[pipe ^ 1][wr0 * 128], 128, 32);
            tdm_load_tile(p.q  + ((size_t)tn * 128 + wr0) * 8,   &sQ  [pipe ^ 1][wr0 * 8],     8, 32);
            tdm_wait(2);                        // current tile's 2 descriptors done
        } else {
            tdm_wait(0);
        }

        // convert this wave's 32 rows f32 -> f16
        for (int i = lane; i < 32 * 32; i += 32) {
            int r = wr0 + (i >> 5), c4 = (i & 31) << 2;
            const float4 f = *reinterpret_cast<const float4*>(&sF32[pipe][r * 128 + c4]);
            h4 hh = { (_Float16)f.x, (_Float16)f.y, (_Float16)f.z, (_Float16)f.w };
            *reinterpret_cast<h4*>(&sA[r * 128 + c4]) = hh;
        }

        // ---- WMMA: k = kv*Wk^T, v = kv*Wv^T, r1pre = kv*Wr1kv^T ----
        // A: 8 ds_load_b128 per 16-row tile; B: registers; 6 independent C chains.
#pragma unroll
        for (int tt = 0; tt < 2; ++tt) {
            const int tb = wr0 + tt * 16;
            v16h Afrag[4];
#pragma unroll
            for (int kt = 0; kt < 4; ++kt) {
                const _Float16* ap = &sA[(tb + lm) * 128 + kt * 32 + kh];
                v8h lo = *(const v8h*)ap;
                v8h hi = *(const v8h*)(ap + 16);
                Afrag[kt] = SHUF16(lo, hi);
            }
            float* Os[3] = { sK, sV, sR1 };
#pragma unroll
            for (int mm = 0; mm < 3; ++mm) {
#pragma unroll
                for (int nt = 0; nt < 2; ++nt) {
                    v8f c = {0.f, 0.f, 0.f, 0.f, 0.f, 0.f, 0.f, 0.f};
#pragma unroll
                    for (int kt = 0; kt < 4; ++kt)
                        c = __builtin_amdgcn_wmma_f32_16x16x32_f16(
                                false, Afrag[kt], false, Bfr[mm][nt][kt],
                                (short)0, c, false, false);
#pragma unroll
                    for (int j = 0; j < 8; ++j)
                        Os[mm][(tb + m0 + j) * 32 + nt * 16 + lm] = c[j];
                }
            }
        }

        // ---- per-row trunk: thread tid = row tid (wave-local). Ordered to keep
        //      live ranges small: res1 first, then q-head, then attention.
        const size_t row = (size_t)t * 128 + tid;

        const float4 q0 = *reinterpret_cast<const float4*>(&sQ[pipe][tid * 8]);
        const float4 q1 = *reinterpret_cast<const float4*>(&sQ[pipe][tid * 8 + 4]);
        float qv[8] = { q0.x, q0.y, q0.z, q0.w, q1.x, q1.y, q1.z, q1.w };

        // (A) res1 = relu(LN(kv@Wr1kv^T + q@Wr1q^T + br1)) @ Wr2^T + br2
        float r1[32];
#pragma unroll
        for (int n = 0; n < 32; ++n) {
            float s = sR1[tid * 32 + n] + sP[O_br1 + n];
#pragma unroll
            for (int j = 0; j < 8; ++j) s += qv[j] * sP[O_Wr1q + n * 8 + j];
            r1[n] = s;
        }
        ln_<32>(r1, &sP[O_gr1], &sP[O_br1n]);
        relu_<32>(r1);
        float res1[16];
        matvec<16, 32>(&sP[O_Wr2], r1, &sP[O_br2], res1);      // r1 dead

        // (B) q head
        float h16[16];
        matvec<16, 8>(&sP[O_Wq1], qv, nullptr, h16);
        ln_<16>(h16, &sP[O_gq], &sP[O_bq]);
        relu_<16>(h16);
        float qh[32];
        matvec<32, 16>(&sP[O_Wq2], h16, &sP[O_bq2], qh);

        // (C) att = softmax(k*qh) in place; x = att*v in place
        float att[32];
        float mx = -3.4e38f;
#pragma unroll
        for (int i = 0; i < 32; ++i) {
            att[i] = (sK[tid * 32 + i] + sP[O_bk + i]) * qh[i];
            mx = fmaxf(mx, att[i]);
        }
        float ssum = 0.f;
#pragma unroll
        for (int i = 0; i < 32; ++i) { att[i] = __expf(att[i] - mx); ssum += att[i]; }
        float inv = 1.0f / ssum;
#pragma unroll
        for (int i = 0; i < 32; ++i)
            att[i] = att[i] * inv * (sV[tid * 32 + i] + sP[O_bv + i]);  // qh dead

        // (D) res2 = relu(LN(x)) @ Wr3^T + br3
        ln_<32>(att, &sP[O_gr2], &sP[O_br2n]);
        relu_<32>(att);
        float res2[16];
        matvec<16, 32>(&sP[O_Wr3], att, &sP[O_br3], res2);     // att dead

        float xb[16], ta[16], tb[16];
        // block1
#pragma unroll
        for (int i = 0; i < 16; ++i) ta[i] = res2[i];
        ln_<16>(ta, &sP[O_g11], &sP[O_b11]); relu_<16>(ta);
        matvec<16, 16>(&sP[O_W11], ta, nullptr, tb);
        ln_<16>(tb, &sP[O_g12], &sP[O_b12]); relu_<16>(tb);
        matvec<16, 16>(&sP[O_W12], tb, nullptr, xb);
        ln_<16>(xb, &sP[O_g13], &sP[O_b13]);
        // block2
#pragma unroll
        for (int i = 0; i < 16; ++i) ta[i] = fmaxf(xb[i] + res1[i], 0.f);
        matvec<16, 16>(&sP[O_W21], ta, nullptr, tb);
        ln_<16>(tb, &sP[O_g21], &sP[O_b21]); relu_<16>(tb);
        matvec<16, 16>(&sP[O_W22], tb, nullptr, xb);
        ln_<16>(xb, &sP[O_g22], &sP[O_b22]);
        // block3
#pragma unroll
        for (int i = 0; i < 16; ++i) ta[i] = fmaxf(xb[i] + res1[i], 0.f);
        matvec<16, 16>(&sP[O_W31], ta, nullptr, tb);
        ln_<16>(tb, &sP[O_g31], &sP[O_b31]); relu_<16>(tb);
        matvec<16, 16>(&sP[O_W32], tb, nullptr, xb);
        ln_<16>(xb, &sP[O_g32], &sP[O_b32]);
        // block4
#pragma unroll
        for (int i = 0; i < 16; ++i) ta[i] = fmaxf(xb[i] + res1[i], 0.f);
        matvec<16, 16>(&sP[O_W41], ta, nullptr, tb);
        ln_<16>(tb, &sP[O_g41], &sP[O_b41]); relu_<16>(tb);
        matvec<16, 16>(&sP[O_W42], tb, &sP[O_b42], xb);

        // final head
        float cat[48];
#pragma unroll
        for (int i = 0; i < 16; ++i) { cat[i] = xb[i]; cat[16 + i] = res1[i]; cat[32 + i] = res2[i]; }
        ln_<48>(cat, &sP[O_gf1], &sP[O_bf1]);
        relu_<48>(cat);
        float h24[24];
        matvec<24, 48>(&sP[O_Wf1], cat, nullptr, h24);
        ln_<24>(h24, &sP[O_gf2], &sP[O_bf2]);
        relu_<24>(h24);
        float o8[8];
        matvec<8, 24>(&sP[O_Wf2], h24, &sP[O_bf2b], o8);

        float4 o0 = { o8[0], o8[1], o8[2], o8[3] };
        float4 o1 = { o8[4], o8[5], o8[6], o8[7] };
        *reinterpret_cast<float4*>(&p.out[row * 8])     = o0;
        *reinterpret_cast<float4*>(&p.out[row * 8 + 4]) = o1;
    }
}

extern "C" void kernel_launch(void* const* d_in, const int* in_sizes, int n_in,
                              void* d_out, int out_size, void* d_ws, size_t ws_size,
                              hipStream_t stream) {
    KParams p;
    int i = 0;
    auto nx = [&]() { return (const float*)d_in[i++]; };
    p.kv = nx();  p.q = nx();
    p.Wk = nx();  p.bk = nx();  p.Wv = nx();  p.bv = nx();
    p.Wq1 = nx(); p.gq = nx();  p.bq = nx();  p.Wq2 = nx(); p.bq2 = nx();
    p.Wr1 = nx(); p.br1 = nx(); p.gr1 = nx(); p.br1n = nx(); p.Wr2 = nx(); p.br2 = nx();
    p.gr2 = nx(); p.br2n = nx(); p.Wr3 = nx(); p.br3 = nx();
    p.g11 = nx(); p.b11 = nx(); p.W11 = nx(); p.g12 = nx(); p.b12 = nx(); p.W12 = nx();
    p.g13 = nx(); p.b13 = nx();
    p.W21 = nx(); p.g21 = nx(); p.b21 = nx(); p.W22 = nx(); p.g22 = nx(); p.b22 = nx();
    p.W31 = nx(); p.g31 = nx(); p.b31 = nx(); p.W32 = nx(); p.g32 = nx(); p.b32 = nx();
    p.W41 = nx(); p.g41 = nx(); p.b41 = nx(); p.W42 = nx(); p.b42 = nx();
    p.gf1 = nx(); p.bf1 = nx(); p.Wf1 = nx(); p.gf2 = nx(); p.bf2 = nx();
    p.Wf2 = nx(); p.bf2b = nx();
    p.out = (float*)d_out;

    const int N = in_sizes[0] / 128;            // 262144 rows
    const int numTiles = N / 128;               // 2048 tiles of 128 rows
    int blocks = numTiles < 512 ? numTiles : 512;
    dim3 grid(blocks), block(128);
    hipLaunchKernelGGL(fused_net, grid, block, 0, stream, p, numTiles);
}
